// GINEncoder_45707041964388
// MI455X (gfx1250) — matrix operand order repaired and checked by reading, not compiled, not run on hardware
//
#include <hip/hip_runtime.h>

typedef __attribute__((ext_vector_type(16))) __bf16 v16bf;
typedef __attribute__((ext_vector_type(8)))  __bf16 v8bf;
typedef __attribute__((ext_vector_type(4)))  __bf16 v4bf;
typedef __attribute__((ext_vector_type(8)))  float  v8f;

#define N_NODES  100000
#define N_EDGES  1600000
#define N_GRAPHS 512
#define DH       256
#define DIN      64
#define BN_EPS   1e-5f
#define MTILE    128

// ---------------------------------------------------------------------------
// Weight conversion: fp32 row-major (K x Nc) -> bf16 column-major padded
// dst[n*KPAD + k] = src[k*Nc + n]
// ---------------------------------------------------------------------------
__global__ void cvt_w(const float* __restrict__ src, __bf16* __restrict__ dst,
                      int K, int Nc, int KPAD) {
    int t = blockIdx.x * blockDim.x + threadIdx.x;
    if (t >= K * Nc) return;
    int n = t / K, k = t % K;
    dst[n * KPAD + k] = (__bf16)src[k * Nc + n];
}

__global__ void zero_f32(float* __restrict__ p, int n) {
    int t = blockIdx.x * blockDim.x + threadIdx.x;
    if (t < n) p[t] = 0.0f;
}

__global__ void copy4(const float4* __restrict__ s, float4* __restrict__ d, int n4) {
    int t = blockIdx.x * blockDim.x + threadIdx.x;
    if (t < n4) d[t] = s[t];
}

// ---------------------------------------------------------------------------
// Edge scatter-add: acc[dst[e]] += x[src[e]]  (acc pre-seeded with x -> h=x+agg)
// One thread handles 4 consecutive channels of one edge; row gathers coalesce;
// atomics resolve in L2 (the 100 MB activation tensor fits in the 192 MB L2).
// ---------------------------------------------------------------------------
__global__ void scatter_add(const float* __restrict__ x, const int* __restrict__ src,
                            const int* __restrict__ dst, float* __restrict__ acc,
                            int nE, int dq, int d) {
    int t = blockIdx.x * blockDim.x + threadIdx.x;
    if (t >= nE * dq) return;
    int e = t / dq, c = (t % dq) * 4;
    int s = src[e], dd = dst[e];
    const float4 v = *(const float4*)(x + (long)s * d + c);
    float* p = acc + (long)dd * d + c;
    atomicAdd(p + 0, v.x);
    atomicAdd(p + 1, v.y);
    atomicAdd(p + 2, v.z);
    atomicAdd(p + 3, v.w);
}

// ---------------------------------------------------------------------------
// WMMA GEMM: C(M x 256) = A(M x KDIM) * B(KDIM x 256) + bias.
// Block tile 128(M) x 64(N), 8 waves; each wave owns a 16-row strip and runs
// 4 x v_wmma_f32_16x16x32_bf16 per K-step (A fragment reused across the four
// 16-col B fragments). B slice (64 cols, full K) LDS-resident; A tile double-
// buffered in LDS -> one barrier per K-step, staging overlaps WMMA.
// EPI 0: relu -> bf16 out.  EPI 1: fp32 out + per-channel sum/sumsq atomics.
// ---------------------------------------------------------------------------
template<int KDIM, bool ABF16, int EPI>
__global__ __launch_bounds__(256) void gemm_wmma(
    const void* __restrict__ Av, const __bf16* __restrict__ Bc,
    const float* __restrict__ bias, void* __restrict__ Outv,
    float* __restrict__ ssum, float* __restrict__ ssq)
{
    constexpr int KPAD = KDIM + 8;
    constexpr int NK   = KDIM / 32;
    __shared__ __align__(16) __bf16 sB[64 * KPAD];
    __shared__ __align__(16) __bf16 sA[2][MTILE * 40];

    const int tid = threadIdx.x;
    const int r0  = blockIdx.x * MTILE;
    const int c0  = blockIdx.y * 64;

    // B slice: contiguous chunk in column-major storage -> flat coalesced copy
    {
        const unsigned int* s = (const unsigned int*)(Bc + (long)c0 * KPAD);
        unsigned int* d = (unsigned int*)sB;
        const int CNT = 64 * KPAD / 2;
        for (int i = tid; i < CNT; i += 256) d[i] = s[i];
    }

    const int wave  = tid >> 5;
    const int lane  = tid & 31;
    const int lhalf = lane >> 4;
    const int l16   = lane & 15;

    v8f acc[4] = {{}, {}, {}, {}};

    // per-lane B base: column l16 (+ j*16 per fragment), K-half per lane-half
    const __bf16* sBw = sB + l16 * KPAD + lhalf * 16;
    const int arow = wave * 16 + l16;

    auto stageA = [&](int ks, int buf) {
        const int k0 = ks * 32;
#pragma unroll
        for (int rc = 0; rc < MTILE / 32; ++rc) {
            int idx = rc * 1024 + tid * 4;
            int row = idx >> 5, k = idx & 31;
            int rg = r0 + row;
            if (rg >= N_NODES) rg = N_NODES - 1;   // tail clamp (rows never stored)
            if (ABF16) {
                const __bf16* Ab = (const __bf16*)Av;
                *(v4bf*)(sA[buf] + row * 40 + k) =
                    *(const v4bf*)(Ab + (long)rg * KDIM + k0 + k);
            } else {
                const float* Af = (const float*)Av;
                const float4 v = *(const float4*)(Af + (long)rg * KDIM + k0 + k);
                v4bf b;
                b[0] = (__bf16)v.x; b[1] = (__bf16)v.y;
                b[2] = (__bf16)v.z; b[3] = (__bf16)v.w;
                *(v4bf*)(sA[buf] + row * 40 + k) = b;
            }
        }
    };

    stageA(0, 0);
    __syncthreads();

    for (int ks = 0; ks < NK; ++ks) {
        if (ks + 1 < NK) stageA(ks + 1, (ks + 1) & 1);   // prefetch next tile

        const int k0 = ks * 32;
        const __bf16* ap = sA[ks & 1] + arow * 40 + lhalf * 16;
        v8bf alo = *(const v8bf*)(ap);
        v8bf ahi = *(const v8bf*)(ap + 8);
        v16bf a = __builtin_shufflevector(alo, ahi,
            0,1,2,3,4,5,6,7,8,9,10,11,12,13,14,15);

#pragma unroll
        for (int j = 0; j < 4; ++j) {
            const __bf16* bp = sBw + j * 16 * KPAD + k0;
            v8bf blo = *(const v8bf*)(bp);
            v8bf bhi = *(const v8bf*)(bp + 8);
            v16bf b = __builtin_shufflevector(blo, bhi,
                0,1,2,3,4,5,6,7,8,9,10,11,12,13,14,15);
            acc[j] = __builtin_amdgcn_wmma_f32_16x16x32_bf16(
                false, a, false, b, (short)0, acc[j], false, false);
        }
        __syncthreads();
    }

    const int rowb = r0 + wave * 16 + lhalf * 8;
#pragma unroll
    for (int j = 0; j < 4; ++j) {
        const int gc = c0 + j * 16 + l16;
        const float bi = bias[gc];
        if (EPI == 0) {
            __bf16* Out = (__bf16*)Outv;
#pragma unroll
            for (int r = 0; r < 8; ++r) {
                const int row = rowb + r;
                if (row < N_NODES) {
                    float v = fmaxf(acc[j][r] + bi, 0.0f);
                    Out[(long)row * DH + gc] = (__bf16)v;
                }
            }
        } else {
            float* Out = (float*)Outv;
            float s = 0.0f, sq = 0.0f;
#pragma unroll
            for (int r = 0; r < 8; ++r) {
                const int row = rowb + r;
                if (row < N_NODES) {
                    float v = acc[j][r] + bi;
                    Out[(long)row * DH + gc] = v;
                    s += v; sq += v * v;
                }
            }
            atomicAdd(&ssum[gc], s);
            atomicAdd(&ssq[gc], sq);
        }
    }
}

// ---------------------------------------------------------------------------
// BatchNorm
// ---------------------------------------------------------------------------
__global__ void bn_finalize(const float* __restrict__ ssum, const float* __restrict__ ssq,
                            float* __restrict__ mu, float* __restrict__ inv) {
    int c = threadIdx.x;
    float m   = ssum[c] * (1.0f / (float)N_NODES);
    float var = ssq[c]  * (1.0f / (float)N_NODES) - m * m;
    mu[c]  = m;
    inv[c] = rsqrtf(var + BN_EPS);
}

__global__ void bn_apply(const float* __restrict__ h2, const float* __restrict__ mu,
                         const float* __restrict__ inv, const float* __restrict__ g,
                         const float* __restrict__ be, float* __restrict__ xcur,
                         float* __restrict__ hsum, int relu, int writeTwo) {
    int t = blockIdx.x * blockDim.x + threadIdx.x;
    if (t >= N_NODES * DH / 4) return;
    int c = (t * 4) & (DH - 1);
    float4 v = ((const float4*)h2)[t];
    float4 o;
    o.x = (v.x - mu[c + 0]) * inv[c + 0] * g[c + 0] + be[c + 0];
    o.y = (v.y - mu[c + 1]) * inv[c + 1] * g[c + 1] + be[c + 1];
    o.z = (v.z - mu[c + 2]) * inv[c + 2] * g[c + 2] + be[c + 2];
    o.w = (v.w - mu[c + 3]) * inv[c + 3] * g[c + 3] + be[c + 3];
    if (relu) {
        o.x = fmaxf(o.x, 0.0f); o.y = fmaxf(o.y, 0.0f);
        o.z = fmaxf(o.z, 0.0f); o.w = fmaxf(o.w, 0.0f);
    }
    ((float4*)xcur)[t] = o;
    if (writeTwo) ((float4*)hsum)[t] = o;   // seeds next layer's h = x + agg
}

// ---------------------------------------------------------------------------
// Global mean pool
// ---------------------------------------------------------------------------
__global__ void pool_count(const int* __restrict__ batch, float* __restrict__ counts) {
    int t = blockIdx.x * blockDim.x + threadIdx.x;
    if (t < N_NODES) atomicAdd(&counts[batch[t]], 1.0f);
}

__global__ void pool_sum(const float* __restrict__ h, const int* __restrict__ batch,
                         float* __restrict__ out) {
    int t = blockIdx.x * blockDim.x + threadIdx.x;
    if (t >= N_NODES * (DH / 4)) return;
    int node = t / (DH / 4), c = (t % (DH / 4)) * 4;
    int g = batch[node];
    float4 v = *(const float4*)(h + (long)node * DH + c);
    float* p = out + g * DH + c;
    atomicAdd(p + 0, v.x);
    atomicAdd(p + 1, v.y);
    atomicAdd(p + 2, v.z);
    atomicAdd(p + 3, v.w);
}

__global__ void pool_div(float* __restrict__ out, const float* __restrict__ counts) {
    int t = blockIdx.x * blockDim.x + threadIdx.x;
    out[t] /= fmaxf(counts[t >> 8], 1.0f);
}

// ---------------------------------------------------------------------------
extern "C" void kernel_launch(void* const* d_in, const int* in_sizes, int n_in,
                              void* d_out, int out_size, void* d_ws, size_t ws_size,
                              hipStream_t stream) {
    (void)in_sizes; (void)n_in; (void)out_size; (void)ws_size;

    const float* x    = (const float*)d_in[0];
    const int*   ei   = (const int*)d_in[1];
    const int*   bat  = (const int*)d_in[2];
    const float* wA[3] = {(const float*)d_in[3],  (const float*)d_in[9],  (const float*)d_in[15]};
    const float* bA[3] = {(const float*)d_in[4],  (const float*)d_in[10], (const float*)d_in[16]};
    const float* wB[3] = {(const float*)d_in[5],  (const float*)d_in[11], (const float*)d_in[17]};
    const float* bB[3] = {(const float*)d_in[6],  (const float*)d_in[12], (const float*)d_in[18]};
    const float* gg[3] = {(const float*)d_in[7],  (const float*)d_in[13], (const float*)d_in[19]};
    const float* bb[3] = {(const float*)d_in[8],  (const float*)d_in[14], (const float*)d_in[20]};
    const int* src = ei;
    const int* dst = ei + N_EDGES;

    char* ws = (char*)d_ws;
    __bf16* WAc[3] = {(__bf16*)(ws + 0),       (__bf16*)(ws + 524288),  (__bf16*)(ws + 1048576)};
    __bf16* WBc[3] = {(__bf16*)(ws + 262144),  (__bf16*)(ws + 786432),  (__bf16*)(ws + 1310720)};
    float* ssum   = (float*)(ws + 1572864);
    float* ssq    = (float*)(ws + 1572864 + 1024);
    float* mu     = (float*)(ws + 1572864 + 2048);
    float* inv    = (float*)(ws + 1572864 + 3072);
    float* counts = (float*)(ws + 1572864 + 4096);
    float*  B1 = (float*)(ws + 2097152);                 // x_cur (N x 256 f32)
    float*  B2 = (float*)(ws + 2097152 + 102400000);     // hsum / h2 (N x 256 f32)
    __bf16* B3 = (__bf16*)(ws + 2097152 + 204800000);    // h1 (N x 256 bf16)

    const int T = 256;
    const dim3 gGemm((N_NODES + MTILE - 1) / MTILE, DH / 64);

    // --- weights -> bf16 column-major padded ---
    cvt_w<<<(DIN * DH + T - 1) / T, T, 0, stream>>>(wA[0], WAc[0], DIN, DH, DIN + 8);
    cvt_w<<<(DH * DH + T - 1) / T, T, 0, stream>>>(wB[0], WBc[0], DH, DH, DH + 8);
    for (int L = 1; L < 3; ++L) {
        cvt_w<<<(DH * DH + T - 1) / T, T, 0, stream>>>(wA[L], WAc[L], DH, DH, DH + 8);
        cvt_w<<<(DH * DH + T - 1) / T, T, 0, stream>>>(wB[L], WBc[L], DH, DH, DH + 8);
    }

    // --- layer 0 (d_in = 64) ---
    copy4<<<(N_NODES * DIN / 4 + T - 1) / T, T, 0, stream>>>(
        (const float4*)x, (float4*)B2, N_NODES * DIN / 4);
    scatter_add<<<(N_EDGES * (DIN / 4) + T - 1) / T, T, 0, stream>>>(
        x, src, dst, B2, N_EDGES, DIN / 4, DIN);
    gemm_wmma<DIN, false, 0><<<gGemm, T, 0, stream>>>(B2, WAc[0], bA[0], B3, nullptr, nullptr);
    zero_f32<<<2, T, 0, stream>>>(ssum, 512);
    gemm_wmma<DH, true, 1><<<gGemm, T, 0, stream>>>(B3, WBc[0], bB[0], B2, ssum, ssq);
    bn_finalize<<<1, DH, 0, stream>>>(ssum, ssq, mu, inv);
    bn_apply<<<(N_NODES * DH / 4 + T - 1) / T, T, 0, stream>>>(
        B2, mu, inv, gg[0], bb[0], B1, B2, 1, 1);
    scatter_add<<<(N_EDGES * (DH / 4) + T - 1) / T, T, 0, stream>>>(
        B1, src, dst, B2, N_EDGES, DH / 4, DH);

    // --- layers 1, 2 ---
    for (int L = 1; L < 3; ++L) {
        gemm_wmma<DH, false, 0><<<gGemm, T, 0, stream>>>(B2, WAc[L], bA[L], B3, nullptr, nullptr);
        zero_f32<<<2, T, 0, stream>>>(ssum, 512);
        gemm_wmma<DH, true, 1><<<gGemm, T, 0, stream>>>(B3, WBc[L], bB[L], B2, ssum, ssq);
        bn_finalize<<<1, DH, 0, stream>>>(ssum, ssq, mu, inv);
        int relu = (L < 2) ? 1 : 0;
        bn_apply<<<(N_NODES * DH / 4 + T - 1) / T, T, 0, stream>>>(
            B2, mu, inv, gg[L], bb[L], B1, B2, relu, relu);
        if (L < 2)
            scatter_add<<<(N_EDGES * (DH / 4) + T - 1) / T, T, 0, stream>>>(
                B1, src, dst, B2, N_EDGES, DH / 4, DH);
    }

    // --- global mean pool ---
    zero_f32<<<(N_GRAPHS * DH + T - 1) / T, T, 0, stream>>>((float*)d_out, N_GRAPHS * DH);
    zero_f32<<<(N_GRAPHS + T - 1) / T, T, 0, stream>>>(counts, N_GRAPHS);
    pool_count<<<(N_NODES + T - 1) / T, T, 0, stream>>>(bat, counts);
    pool_sum<<<(N_NODES * (DH / 4) + T - 1) / T, T, 0, stream>>>(B1, bat, (float*)d_out);
    pool_div<<<(N_GRAPHS * DH) / T, T, 0, stream>>>((float*)d_out, counts);
}
